// LSTM_8057358648149
// MI455X (gfx1250) — matrix operand order repaired and checked by reading, not compiled, not run on hardware
//
#include <hip/hip_runtime.h>
#include <hip/hip_bf16.h>

// LSTM: T=2048, B=2048, I=1, H=20.  One wave (32 lanes) owns 16 batch rows and
// iterates all T steps with h,c resident in VGPRs.  Per-step GEMM
// z(80x16) = U_aug(80x22) @ [h;1;x](22x16) done with v_wmma_f32_16x16x32_f16,
// gates padded per-gate to 32 rows (8 M-tiles), K padded to 32 (one WMMA K-chunk).
// Ootomo f16-split (hi/lo) keeps ~f32 accuracy: 3 WMMAs per tile.

#define TT 2048
#define BB 2048
#define HH 20

typedef __attribute__((ext_vector_type(16))) _Float16 v16h;
typedef __attribute__((ext_vector_type(8)))  float    v8f;
typedef __attribute__((ext_vector_type(2)))  _Float16 h2t;

union V16 { v16h v; h2t p[8]; unsigned u[8]; };

__device__ __forceinline__ float fast_tanh(float x) {
#if __has_builtin(__builtin_amdgcn_tanhf)
  return __builtin_amdgcn_tanhf(x);
#elif __has_builtin(__builtin_amdgcn_tanh_f32)
  return __builtin_amdgcn_tanh_f32(x);
#else
  // tanh(x) = 1 - 2/(exp(2x)+1);  exp(2x) = exp2(2*log2(e)*x)
  float e = __builtin_amdgcn_exp2f(2.8853900817779268f * x);
  return 1.0f - 2.0f * __builtin_amdgcn_rcpf(e + 1.0f);
#endif
}
__device__ __forceinline__ float fast_sigmoid(float x) {
  return __builtin_fmaf(0.5f, fast_tanh(0.5f * x), 0.5f);
}

// SWAPX16: swap lanes 0-15 <-> 16-31 (group-of-32 ds_swizzle, xor=0x10, and=0x1f)
__device__ __forceinline__ unsigned swap16(unsigned v) {
  return (unsigned)__builtin_amdgcn_ds_swizzle((int)v, 0x401f);
}

// Split two f32 into packed f16 hi + f16 residual lo.
__device__ __forceinline__ void pack2(float a, float b, unsigned& hi, unsigned& lo) {
  _Float16 ah = (_Float16)a, bh = (_Float16)b;
  h2t hv = {ah, bh};
  h2t lv = {(_Float16)(a - (float)ah), (_Float16)(b - (float)bh)};
  hi = __builtin_bit_cast(unsigned, hv);
  lo = __builtin_bit_cast(unsigned, lv);
}

__global__ void __launch_bounds__(32)
lstm_wmma_kernel(const float* __restrict__ x, const float* __restrict__ W,
                 const float* __restrict__ U, const float* __restrict__ bw,
                 const float* __restrict__ bu, float* __restrict__ out) {
  const int lane = threadIdx.x;     // 0..31 (wave32)
  const int bl   = lane & 15;       // batch column within tile
  const int hsel = lane >> 4;       // lane half
  const int b0   = blockIdx.x * 16; // first batch row of this wave

  // ---------------- Build A fragments: U_aug = [U | (bw+bu) | W | 0..] ----------------
  // Padded row p = 32*gate + q (q<20 valid).  A K order used by this code (contiguous):
  // vgpr s: K = 2s+{0,1} + 16*half.
  v16h Ahi[8], Alo[8];
  {
    const int prow = 16 * 0 + bl; (void)prow;
#pragma unroll
    for (int tl = 0; tl < 8; ++tl) {
      int p = 16 * tl + bl;
      int g = p >> 5, q = p & 31;
      int row = g * 20 + q;                // U row (valid iff q<20)
      V16 hi, lo;
#pragma unroll
      for (int s = 0; s < 8; ++s) {
        int kb = 2 * s + 16 * hsel;
        float a0 = 0.0f, a1 = 0.0f;
        if (q < 20) {
          int k0 = kb, k1 = kb + 1;
          a0 = (k0 < 20) ? U[row * 20 + k0]
                         : (k0 == 20 ? (bw[row] + bu[row]) : (k0 == 21 ? W[row] : 0.0f));
          a1 = (k1 < 20) ? U[row * 20 + k1]
                         : (k1 == 20 ? (bw[row] + bu[row]) : (k1 == 21 ? W[row] : 0.0f));
        }
        _Float16 f0 = (_Float16)a0, f1 = (_Float16)a1;
        hi.p[s] = h2t{f0, f1};
        lo.p[s] = h2t{(_Float16)(a0 - (float)f0), (_Float16)(a1 - (float)f1)};
      }
      Ahi[tl] = hi.v;
      Alo[tl] = lo.v;
    }
  }

  // ---------------- Recurrent state (D-layout: lane = batch + 16*half, q = v + 8*half) ----
  float ht0[8], ct0[8];   // q = 0..15
  float ht1[4], ct1[4];   // q = 16..19 (valid in lanes 0-15)
#pragma unroll
  for (int v = 0; v < 8; ++v) { ht0[v] = 0.0f; ct0[v] = 0.0f; }
#pragma unroll
  for (int v = 0; v < 4; ++v) { ht1[v] = 0.0f; ct1[v] = 0.0f; }

  const v8f zz = {0.f, 0.f, 0.f, 0.f, 0.f, 0.f, 0.f, 0.f};
  const bool upper = (hsel != 0);

  const float* xw = x + b0 + bl;                       // x[t*BB + b0+bl]  (I==1)
  const size_t off0 = (size_t)(b0 + bl) * HH + 8 * hsel;
  const size_t off1 = (size_t)(b0 + bl) * HH + 16;

#pragma unroll 1
  for (int t = 0; t < TT; ++t) {
    float xv = xw[(size_t)t * BB];
    if (t + 32 < TT) __builtin_prefetch(xw + (size_t)(t + 32) * BB, 0, 0);

    // ---- build B = [h ; 1 ; x ; 0..] (32x16 f16), hi + residual lo ----
    unsigned pa, pal, pb, pbl, pc, pcl, pd, pdl, pe, pel, pf, pfl, px, pxl;
    pack2(ht0[0], ht0[1], pa, pal);   // (h0,h1)|(h8,h9)
    pack2(ht0[2], ht0[3], pb, pbl);   // (h2,h3)|(h10,h11)
    pack2(ht0[4], ht0[5], pc, pcl);   // (h4,h5)|(h12,h13)
    pack2(ht0[6], ht0[7], pd, pdl);   // (h6,h7)|(h14,h15)
    pack2(ht1[0], ht1[1], pe, pel);   // (h16,h17)|garbage
    pack2(ht1[2], ht1[3], pf, pfl);   // (h18,h19)|garbage
    pack2(1.0f,   xv,     px, pxl);   // (1,x) ; lo = (0, x_lo)

    unsigned sa = swap16(pa),  sb = swap16(pb),  sc = swap16(pc),  sd = swap16(pd);
    unsigned se = swap16(pe),  sf = swap16(pf);
    unsigned sal = swap16(pal), sbl = swap16(pbl), scl = swap16(pcl), sdl = swap16(pdl);
    unsigned sel = swap16(pel), sfl = swap16(pfl);

    // B K order used by this code (same as A): vgpr v holds K = 2v+{0,1} in lanes 0-15, K = 16+2v+{0,1} in 16-31.
    V16 Bh, Bl;
    Bh.u[0] = upper ? se : pa;   Bl.u[0] = upper ? sel : pal;   // K0,1 | K16,17
    Bh.u[1] = upper ? sf : pb;   Bl.u[1] = upper ? sfl : pbl;   // K2,3 | K18,19
    Bh.u[2] = upper ? px : pc;   Bl.u[2] = upper ? pxl : pcl;   // K4,5 | K20(=1),21(=x)
    Bh.u[3] = upper ? 0u : pd;   Bl.u[3] = upper ? 0u  : pdl;   // K6,7 | 0
    Bh.u[4] = upper ? 0u : sa;   Bl.u[4] = upper ? 0u  : sal;   // K8,9 | 0
    Bh.u[5] = upper ? 0u : sb;   Bl.u[5] = upper ? 0u  : sbl;   // K10,11 | 0
    Bh.u[6] = upper ? 0u : sc;   Bl.u[6] = upper ? 0u  : scl;   // K12,13 | 0
    Bh.u[7] = upper ? 0u : sd;   Bl.u[7] = upper ? 0u  : sdl;   // K14,15 | 0

    // ---- z = Ulo*Bhi + Uhi*Blo + Uhi*Bhi  (8 independent tiles) ----
    v8f z[8];
#pragma unroll
    for (int tl = 0; tl < 8; ++tl) {
      v8f acc = __builtin_amdgcn_wmma_f32_16x16x32_f16(false, Alo[tl], false, Bh.v,
                                                       (short)0, zz, false, false);
      acc = __builtin_amdgcn_wmma_f32_16x16x32_f16(false, Ahi[tl], false, Bl.v,
                                                   (short)0, acc, false, false);
      z[tl] = __builtin_amdgcn_wmma_f32_16x16x32_f16(false, Ahi[tl], false, Bh.v,
                                                     (short)0, acc, false, false);
    }

    // ---- gates + cell update (tiles 0/1=f, 2/3=i, 4/5=o, 6/7=g), pure elementwise ----
#pragma unroll
    for (int v = 0; v < 8; ++v) {
      float f = fast_sigmoid(z[0][v]);
      float i = fast_sigmoid(z[2][v]);
      float o = fast_sigmoid(z[4][v]);
      float g = fast_tanh(z[6][v]);
      float c = __builtin_fmaf(f, ct0[v], i * g);
      ct0[v] = c;
      ht0[v] = o * fast_tanh(c);
    }
#pragma unroll
    for (int v = 0; v < 4; ++v) {
      float f = fast_sigmoid(z[1][v]);
      float i = fast_sigmoid(z[3][v]);
      float o = fast_sigmoid(z[5][v]);
      float g = fast_tanh(z[7][v]);
      float c = __builtin_fmaf(f, ct1[v], i * g);
      ct1[v] = c;
      ht1[v] = o * fast_tanh(c);
    }

    // ---- write h_seq[t] (16B-aligned float4 stores) ----
    float* ot = out + (size_t)t * (BB * HH);
    float4 s0 = {ht0[0], ht0[1], ht0[2], ht0[3]};
    float4 s1 = {ht0[4], ht0[5], ht0[6], ht0[7]};
    *(float4*)(ot + off0)     = s0;
    *(float4*)(ot + off0 + 4) = s1;
    if (lane < 16) {
      float4 s2 = {ht1[0], ht1[1], ht1[2], ht1[3]};
      *(float4*)(ot + off1) = s2;
    }
  }

  // ---- h_last, c_last ----
  float* hl = out + (size_t)TT * BB * HH;
  float* cl = hl + (size_t)BB * HH;
  {
    float4 a0 = {ht0[0], ht0[1], ht0[2], ht0[3]};
    float4 a1 = {ht0[4], ht0[5], ht0[6], ht0[7]};
    float4 c0 = {ct0[0], ct0[1], ct0[2], ct0[3]};
    float4 c1 = {ct0[4], ct0[5], ct0[6], ct0[7]};
    *(float4*)(hl + off0)     = a0;
    *(float4*)(hl + off0 + 4) = a1;
    *(float4*)(cl + off0)     = c0;
    *(float4*)(cl + off0 + 4) = c1;
    if (lane < 16) {
      float4 a2 = {ht1[0], ht1[1], ht1[2], ht1[3]};
      float4 c2 = {ct1[0], ct1[1], ct1[2], ct1[3]};
      *(float4*)(hl + off1) = a2;
      *(float4*)(cl + off1) = c2;
    }
  }
}

extern "C" void kernel_launch(void* const* d_in, const int* in_sizes, int n_in,
                              void* d_out, int out_size, void* d_ws, size_t ws_size,
                              hipStream_t stream) {
  const float* x  = (const float*)d_in[0];
  const float* W  = (const float*)d_in[1];
  const float* U  = (const float*)d_in[2];
  const float* bw = (const float*)d_in[3];
  const float* bu = (const float*)d_in[4];
  float* out = (float*)d_out;
  (void)in_sizes; (void)n_in; (void)out_size; (void)d_ws; (void)ws_size;

  lstm_wmma_kernel<<<dim3(BB / 16), dim3(32), 0, stream>>>(x, W, U, bw, bu, out);
}
